// MLP_52450140619487
// MI455X (gfx1250) — matrix operand order repaired and checked by reading, not compiled, not run on hardware
//
#include <hip/hip_runtime.h>

// ---------------------------------------------------------------------------
// MI455X / gfx1250 implementation.
// GEMM-heavy graph-MLP + BiLSTM: all matmuls on v_wmma_f32_16x16x32_f16,
// recurrent weights staged to LDS (async-to-LDS path when available),
// edge features gathered on the fly from an L2-resident node table.
// ---------------------------------------------------------------------------

#define AS1 __attribute__((address_space(1)))
#define AS3 __attribute__((address_space(3)))

typedef __attribute__((ext_vector_type(16))) _Float16 v16h;
typedef __attribute__((ext_vector_type(8)))  _Float16 v8h;
typedef __attribute__((ext_vector_type(8)))  float    v8f;
typedef int v4i_vs __attribute__((vector_size(16)));   // matches builtin param type

#ifndef __has_builtin
#define __has_builtin(x) 0
#endif
#if __has_builtin(__builtin_amdgcn_global_load_async_to_lds_b128)
#define HAVE_ASYNC_LDS 1
#else
#define HAVE_ASYNC_LDS 0
#endif

// Problem constants (fixed by the reference).
#define NN        2048      // batch rows / graph nodes
#define SEQ       32        // sequence length
#define EMB       128       // embedding / hidden dim
#define GATES     512       // 4*EMB
#define HID       512
#define NEDGE     32768
#define NODE_DIM  544       // 4*EMB + 32
#define HCAT      256       // 2*EMB
#define EDGE_K    1088      // 2*NODE_DIM

// ---------------- WMMA helpers -------------------------------------------

__device__ __forceinline__ v8f wmma32(v16h a, v16h b, v8f c) {
  // D = A(16x32 f16) * B(32x16 f16) + C(16x16 f32)
  return __builtin_amdgcn_wmma_f32_16x16x32_f16(
      /*neg_a=*/false, a, /*neg_b=*/false, b,
      /*c_mod=*/(short)0, c, /*reuse_a=*/false, /*reuse_b=*/false);
}

// A fragment, row-major A. rowp = &A[row_for_this_lane][0] (row = m0 + (lane&15)).
// ISA 16-bit A layout: halves 0..7 hold K = k0 + 8*hi + (0..7),
//                      halves 8..15 hold K = k0 + 16 + 8*hi + (0..7).
__device__ __forceinline__ v16h load_a_frag(const _Float16* rowp, int k0, int hi) {
  const _Float16* p = rowp + k0 + 8 * hi;
  v8h lo = *(const v8h*)p;
  v8h hh = *(const v8h*)(p + 16);
  return __builtin_shufflevector(lo, hh, 0,1,2,3,4,5,6,7,8,9,10,11,12,13,14,15);
}

// B fragment from transposed weights Bt[N][K] (ldb == K).
// ISA 16-bit B layout: lane l -> n = n0 + (l&15), halves 0..7 hold K = k0 + 8*hi + j, halves 8..15 hold K = k0 + 16 + 8*hi + (j-8).
__device__ __forceinline__ v16h load_b_frag(const _Float16* Bt, int ldb,
                                            int n0, int k0, int lane) {
  int n  = n0 + (lane & 15);
  int kb = k0 + 8 * (lane >> 4);
  const _Float16* p = Bt + (size_t)n * ldb + kb;
  v8h lo = *(const v8h*)p;
  v8h hh = *(const v8h*)(p + 16);
  return __builtin_shufflevector(lo, hh, 0,1,2,3,4,5,6,7,8,9,10,11,12,13,14,15);
}

__device__ __forceinline__ float leaky_f(float v) { return v >= 0.f ? v : 0.01f * v; }
__device__ __forceinline__ float sigmoid_f(float v) { return 1.f / (1.f + __expf(-v)); }

// ---------------- Prep kernels -------------------------------------------

// dst[c][r] = (f16) src[r][c]   (weights -> [N][K] f16 for B-fragment loads)
__global__ void k_transpose_f16(const float* __restrict__ src,
                                _Float16* __restrict__ dst, int R, int C) {
  int idx = blockIdx.x * blockDim.x + threadIdx.x;
  if (idx >= R * C) return;
  int r = idx / C, c = idx - r * C;
  dst[(size_t)c * R + r] = (_Float16)src[(size_t)r * C + c];
}

// url_h[n][t][k] = (f16) embed_url[inputs_s[n][t]][k]
__global__ void k_gather_url(const int* __restrict__ s,
                             const float* __restrict__ emb,
                             _Float16* __restrict__ out) {
  int idx = blockIdx.x * blockDim.x + threadIdx.x;
  if (idx >= NN * SEQ * EMB) return;
  int k  = idx & (EMB - 1);
  int nt = idx >> 7;
  int t  = nt & (SEQ - 1);
  int n  = nt >> 5;
  out[idx] = (_Float16)emb[(size_t)s[n * SEQ + t] * EMB + k];
}

// node[n][128..543] = concat(cat_emb, country_emb, sl_emb, ip) as f16
__global__ void k_node_static(const int* __restrict__ c, const int* __restrict__ co,
                              const int* __restrict__ sl, const float* __restrict__ ip,
                              const float* __restrict__ ecat, const float* __restrict__ ecou,
                              const float* __restrict__ esl, _Float16* __restrict__ node) {
  int idx = blockIdx.x * blockDim.x + threadIdx.x;
  if (idx >= NN * 416) return;
  int n = idx / 416, j = idx - n * 416;
  float v;
  if (j < 128)      v = ecat[(size_t)c[n]  * 128 + j];
  else if (j < 256) v = ecou[(size_t)co[n] * 128 + (j - 128)];
  else if (j < 384) v = esl[(size_t)sl[n]  * 128 + (j - 256)];
  else              v = ip[(size_t)n * 32 + (j - 384)];
  node[(size_t)n * NODE_DIM + 128 + j] = (_Float16)v;
}

// ---------------- LSTM kernel --------------------------------------------
// One workgroup (8 wave32) owns 16 batch rows; W_hh (128 KB f16) lives in LDS,
// h/c/z live in LDS, 32 timesteps swept locally (rows are independent LSTMs).
// Per step: z[16][512] = X_t@W_ih + h@W_hh via 256 WMMAs spread over 8 waves.

#define LSTM_LDS_BYTES (GATES*EMB*2 /*whh*/ + 16*GATES*4 /*z*/ + \
                        16*EMB*4*2 /*h,c f32*/ + 16*EMB*2 /*h f16*/)

__global__ void __launch_bounds__(256) k_lstm(
    const _Float16* __restrict__ urlh,   // [NN][SEQ][EMB] f16
    const int* __restrict__ mask,        // [NN][SEQ]
    const _Float16* __restrict__ wih_t,  // [GATES][EMB] f16
    const _Float16* __restrict__ whh_t,  // [GATES][EMB] f16
    const float* __restrict__ bias,      // [GATES]
    _Float16* __restrict__ hcat,         // [NN][HCAT] f16
    int reverse, int col_off) {
  extern __shared__ char smem[];
  _Float16* lds_whh = (_Float16*)smem;                          // 512*128 f16
  float*    lds_z   = (float*)(smem + GATES * EMB * 2);         // 16*512 f32
  float*    lds_h   = (float*)((char*)lds_z + 16 * GATES * 4);  // 16*128 f32
  float*    lds_c   = lds_h + 16 * EMB;                         // 16*128 f32
  _Float16* lds_h16 = (_Float16*)(lds_c + 16 * EMB);            // 16*128 f16

  const int tid  = threadIdx.x;
  const int lane = tid & 31;
  const int wave = tid >> 5;
  const int hi   = lane >> 4;
  const int row0 = blockIdx.x * 16;

  // Stage W_hh^T into LDS (async-to-LDS when the toolchain exposes it).
#if HAVE_ASYNC_LDS
  {
    char* g = (char*)whh_t;
    char* l = (char*)lds_whh;
    for (int off = tid * 16; off < GATES * EMB * 2; off += 256 * 16) {
      __builtin_amdgcn_global_load_async_to_lds_b128(
          (AS1 v4i_vs*)(g + off), (AS3 v4i_vs*)(l + off), 0, 0);
    }
    asm volatile("s_wait_asynccnt 0" ::: "memory");
  }
#else
  for (int off = tid * 8; off < GATES * EMB; off += 256 * 8)
    *(v8h*)(lds_whh + off) = *(const v8h*)(whh_t + off);
#endif
  for (int i = tid; i < 16 * EMB; i += 256) {
    lds_h[i] = 0.f; lds_c[i] = 0.f; lds_h16[i] = (_Float16)0.f;
  }
  __syncthreads();

  for (int tt = 0; tt < SEQ; ++tt) {
    const int t = reverse ? (SEQ - 1 - tt) : tt;
    const _Float16* xrow = urlh + ((size_t)(row0 + (lane & 15)) * SEQ + t) * EMB;
    const _Float16* hrow = lds_h16 + (lane & 15) * EMB;

    // Each wave computes 4 N-tiles (64 of the 512 gate columns).
    for (int i = 0; i < 4; ++i) {
      const int n0 = (wave * 4 + i) * 16;
      v8f acc = {};
#pragma unroll
      for (int k0 = 0; k0 < EMB; k0 += 32) {
        v16h ax = load_a_frag(xrow, k0, hi);
        v16h bw = load_b_frag(wih_t, EMB, n0, k0, lane);
        acc = wmma32(ax, bw, acc);
        v16h ah = load_a_frag(hrow, k0, hi);
        v16h bu = load_b_frag(lds_whh, EMB, n0, k0, lane);
        acc = wmma32(ah, bu, acc);
      }
      const int n = n0 + (lane & 15);
#pragma unroll
      for (int r = 0; r < 8; ++r)
        lds_z[(r + 8 * hi) * GATES + n] = acc[r];
    }
    __syncthreads();

    // Gate nonlinearity + masked state update: thread -> (row m, 8 cols).
    {
      const int m  = tid >> 4;
      const int jb = (tid & 15) * 8;
      const int mt = mask[(row0 + m) * SEQ + t];
#pragma unroll
      for (int q = 0; q < 8; ++q) {
        const int j = jb + q;
        float zi = lds_z[m * GATES + j]        + bias[j];
        float zf = lds_z[m * GATES + 128 + j]  + bias[128 + j];
        float zg = lds_z[m * GATES + 256 + j]  + bias[256 + j];
        float zo = lds_z[m * GATES + 384 + j]  + bias[384 + j];
        float ig = sigmoid_f(zi), fg = sigmoid_f(zf);
        float gg = tanhf(zg),     og = sigmoid_f(zo);
        float c_old = lds_c[m * EMB + j];
        float h_old = lds_h[m * EMB + j];
        float cn = fg * c_old + ig * gg;
        float hn = og * tanhf(cn);
        float hN = mt ? hn : h_old;
        float cN = mt ? cn : c_old;
        lds_h[m * EMB + j] = hN;
        lds_c[m * EMB + j] = cN;
        lds_h16[m * EMB + j] = (_Float16)hN;
      }
    }
    __syncthreads();
  }

  for (int i = tid; i < 16 * EMB; i += 256) {
    int m = i >> 7, j = i & 127;
    hcat[(size_t)(row0 + m) * HCAT + col_off + j] = (_Float16)lds_h[i];
  }
}

// ---------------- Generic f16 WMMA GEMM (+bias, +leaky) -------------------
// One wave per 16x16 output tile; Bt is pre-transposed [N][K].
__global__ void __launch_bounds__(256) k_gemm(
    const _Float16* __restrict__ A, int lda,
    const _Float16* __restrict__ Bt, int ldb,  // ldb == K
    const float* __restrict__ bias,
    _Float16* __restrict__ C, int ldc, int col_off,
    int M, int Ncols, int K, int leaky) {
  const int lane = threadIdx.x & 31, wave = threadIdx.x >> 5, hi = lane >> 4;
  const int ntiles = Ncols >> 4;
  const int tile = blockIdx.x * 8 + wave;
  if (tile >= (M >> 4) * ntiles) return;
  const int mt = tile / ntiles, nt = tile - mt * ntiles;
  const int m0 = mt << 4, n0 = nt << 4;
  const _Float16* arow = A + (size_t)(m0 + (lane & 15)) * lda;
  v8f acc = {};
  for (int k0 = 0; k0 < K; k0 += 32) {
    v16h a = load_a_frag(arow, k0, hi);
    v16h b = load_b_frag(Bt, ldb, n0, k0, lane);
    acc = wmma32(a, b, acc);
  }
  const int n = n0 + (lane & 15);
  const float bv = bias ? bias[n] : 0.f;
#pragma unroll
  for (int r = 0; r < 8; ++r) {
    float v = acc[r] + bv;
    if (leaky) v = leaky_f(v);
    C[(size_t)(m0 + r + 8 * hi) * ldc + col_off + n] = (_Float16)v;
  }
}

// ---------------- Edge layer-1 GEMM with on-the-fly gather ----------------
// A row e = concat(node[src[e]], node[dst[e]]) (K=1088), never materialized.
// 544 % 32 == 0 so every 32-wide K fragment lies wholly in one half.
__global__ void __launch_bounds__(256) k_edge_gemm(
    const _Float16* __restrict__ node,  // [NN][NODE_DIM] f16 (L2 resident)
    const int* __restrict__ src, const int* __restrict__ dst,
    const _Float16* __restrict__ W1t,   // [HID][EDGE_K] f16
    const float* __restrict__ bias,
    _Float16* __restrict__ C) {         // [NEDGE][HID] f16
  const int lane = threadIdx.x & 31, wave = threadIdx.x >> 5, hi = lane >> 4;
  const int ntiles = HID >> 4;
  const int tile = blockIdx.x * 8 + wave;
  if (tile >= (NEDGE >> 4) * ntiles) return;
  const int mt = tile / ntiles, nt = tile - mt * ntiles;
  const int e  = (mt << 4) + (lane & 15);
  const int n0 = nt << 4;
  const _Float16* srow = node + (size_t)src[e] * NODE_DIM;
  const _Float16* drow = node + (size_t)dst[e] * NODE_DIM;
  v8f acc = {};
  for (int k0 = 0; k0 < EDGE_K; k0 += 32) {
    const _Float16* arow = (k0 < NODE_DIM) ? srow : (drow - NODE_DIM);
    v16h a = load_a_frag(arow, k0, hi);
    v16h b = load_b_frag(W1t, EDGE_K, n0, k0, lane);
    acc = wmma32(a, b, acc);
  }
  const int n = n0 + (lane & 15);
  const float bv = bias[n];
#pragma unroll
  for (int r = 0; r < 8; ++r) {
    float v = leaky_f(acc[r] + bv);
    C[(size_t)((mt << 4) + r + 8 * hi) * HID + n] = (_Float16)v;
  }
}

// ---------------- Final 512 -> 2 projection (VALU, tiny) ------------------
__global__ void k_out(const _Float16* __restrict__ x2, const float* __restrict__ W3,
                      const float* __restrict__ b3, float* __restrict__ out) {
  int idx = blockIdx.x * blockDim.x + threadIdx.x;
  if (idx >= NEDGE * 2) return;
  int e = idx >> 1, j = idx & 1;
  const _Float16* xr = x2 + (size_t)e * HID;
  __builtin_prefetch(xr + HID, 0, 0);
  float acc = b3[j];
#pragma unroll 8
  for (int k = 0; k < HID; ++k) acc += (float)xr[k] * W3[k * 2 + j];
  out[idx] = acc;
}

// ---------------- Host orchestration -------------------------------------

extern "C" void kernel_launch(void* const* d_in, const int* in_sizes, int n_in,
                              void* d_out, int out_size, void* d_ws, size_t ws_size,
                              hipStream_t stream) {
  (void)in_sizes; (void)n_in; (void)out_size; (void)ws_size;
  const int*   inputs_s  = (const int*)d_in[0];
  const int*   inputs_sm = (const int*)d_in[1];
  const int*   inputs_c  = (const int*)d_in[2];
  const int*   inputs_co = (const int*)d_in[3];
  const int*   inputs_sl = (const int*)d_in[4];
  const float* inputs_ip = (const float*)d_in[5];
  const int*   src       = (const int*)d_in[6];
  const int*   dst       = (const int*)d_in[7];
  const float* embed_url = (const float*)d_in[8];
  const float* embed_cat = (const float*)d_in[9];
  const float* embed_cou = (const float*)d_in[10];
  const float* embed_sl  = (const float*)d_in[11];
  const float* W_ih_f = (const float*)d_in[12];
  const float* W_hh_f = (const float*)d_in[13];
  const float* b_f    = (const float*)d_in[14];
  const float* W_ih_b = (const float*)d_in[15];
  const float* W_hh_b = (const float*)d_in[16];
  const float* b_b    = (const float*)d_in[17];
  const float* fcW    = (const float*)d_in[18];
  const float* fcb    = (const float*)d_in[19];
  const float* W1     = (const float*)d_in[20];
  const float* b1     = (const float*)d_in[21];
  const float* W2     = (const float*)d_in[22];
  const float* b2     = (const float*)d_in[23];
  const float* W3     = (const float*)d_in[24];
  const float* b3     = (const float*)d_in[25];
  float* out = (float*)d_out;

  // Workspace carve-out (256B aligned).
  char* ws = (char*)d_ws;
  size_t off = 0;
  auto carve = [&](size_t bytes) -> char* {
    char* p = ws + off;
    off = (off + bytes + 255) & ~(size_t)255;
    return p;
  };
  _Float16* wihf_t = (_Float16*)carve((size_t)GATES * EMB * 2);
  _Float16* whhf_t = (_Float16*)carve((size_t)GATES * EMB * 2);
  _Float16* wihb_t = (_Float16*)carve((size_t)GATES * EMB * 2);
  _Float16* whhb_t = (_Float16*)carve((size_t)GATES * EMB * 2);
  _Float16* fcw_t  = (_Float16*)carve((size_t)EMB * HCAT * 2);
  _Float16* w1_t   = (_Float16*)carve((size_t)HID * EDGE_K * 2);
  _Float16* w2_t   = (_Float16*)carve((size_t)HID * HID * 2);
  _Float16* url_h  = (_Float16*)carve((size_t)NN * SEQ * EMB * 2);
  _Float16* hcat_h = (_Float16*)carve((size_t)NN * HCAT * 2);
  _Float16* node_h = (_Float16*)carve((size_t)NN * NODE_DIM * 2);
  _Float16* x1     = (_Float16*)carve((size_t)NEDGE * HID * 2);
  _Float16* x2     = (_Float16*)carve((size_t)NEDGE * HID * 2);

  const int TB = 256;
  auto cdiv = [](int a, int b) { return (a + b - 1) / b; };

  // 1) Weight transposes (f32 -> f16, [N][K]).
  k_transpose_f16<<<cdiv(128 * 512, TB),  TB, 0, stream>>>(W_ih_f, wihf_t, 128, 512);
  k_transpose_f16<<<cdiv(128 * 512, TB),  TB, 0, stream>>>(W_hh_f, whhf_t, 128, 512);
  k_transpose_f16<<<cdiv(128 * 512, TB),  TB, 0, stream>>>(W_ih_b, wihb_t, 128, 512);
  k_transpose_f16<<<cdiv(128 * 512, TB),  TB, 0, stream>>>(W_hh_b, whhb_t, 128, 512);
  k_transpose_f16<<<cdiv(256 * 128, TB),  TB, 0, stream>>>(fcW,    fcw_t,  256, 128);
  k_transpose_f16<<<cdiv(1088 * 512, TB), TB, 0, stream>>>(W1,     w1_t,  1088, 512);
  k_transpose_f16<<<cdiv(512 * 512, TB),  TB, 0, stream>>>(W2,     w2_t,   512, 512);

  // 2) Gathers.
  k_gather_url<<<cdiv(NN * SEQ * EMB, TB), TB, 0, stream>>>(inputs_s, embed_url, url_h);
  k_node_static<<<cdiv(NN * 416, TB), TB, 0, stream>>>(
      inputs_c, inputs_co, inputs_sl, inputs_ip, embed_cat, embed_cou, embed_sl, node_h);

  // 3) BiLSTM: 128 workgroups x 16 rows, W_hh cached in LDS (184 KB dyn LDS).
  k_lstm<<<NN / 16, TB, LSTM_LDS_BYTES, stream>>>(
      url_h, inputs_sm, wihf_t, whhf_t, b_f, hcat_h, /*reverse=*/0, /*col=*/0);
  k_lstm<<<NN / 16, TB, LSTM_LDS_BYTES, stream>>>(
      url_h, inputs_sm, wihb_t, whhb_t, b_b, hcat_h, /*reverse=*/1, /*col=*/EMB);

  // 4) fc_lstm: [2048,256]x[256,128] + leaky -> node_h[:,0:128].
  k_gemm<<<(NN / 16) * (EMB / 16) / 8, TB, 0, stream>>>(
      hcat_h, HCAT, fcw_t, HCAT, fcb, node_h, NODE_DIM, 0, NN, EMB, HCAT, 1);

  // 5) Edge MLP.
  k_edge_gemm<<<(NEDGE / 16) * (HID / 16) / 8, TB, 0, stream>>>(
      node_h, src, dst, w1_t, b1, x1);
  k_gemm<<<(NEDGE / 16) * (HID / 16) / 8, TB, 0, stream>>>(
      x1, HID, w2_t, HID, b2, x2, HID, 0, NEDGE, HID, HID, 1);
  k_out<<<cdiv(NEDGE * 2, TB), TB, 0, stream>>>(x2, W3, b3, out);
}
